// SparseAtlasMoE_3066606649537
// MI455X (gfx1250) — compile-verified, matrix operand order checked by reading, not run on hardware
//
#include <hip/hip_runtime.h>

#define BATCH 4096
#define ATL   16
#define DIM   512
#define DIM2  1024
#define HEADS 8
#define BT    (BATCH*ATL)   // 65536 tokens

typedef __bf16 bf16;
typedef bf16  bf16x16 __attribute__((ext_vector_type(16)));
typedef float f32x8   __attribute__((ext_vector_type(8)));
typedef unsigned int u32x4 __attribute__((ext_vector_type(4)));

union Frag { bf16x16 v; u32x4 u[2]; };

__device__ __forceinline__ bf16 f2bf(float f){
  unsigned u = __builtin_bit_cast(unsigned, f);
  unsigned r = u + 0x7FFFu + ((u >> 16) & 1u);
  unsigned short h = (unsigned short)(r >> 16);
  return __builtin_bit_cast(bf16, h);
}
__device__ __forceinline__ float bf2f(bf16 x){
  unsigned short h = __builtin_bit_cast(unsigned short, x);
  unsigned u = ((unsigned)h) << 16;
  return __builtin_bit_cast(float, u);
}

// low 32 bits of a generic pointer to __shared__ == LDS byte offset on AMDGPU
__device__ __forceinline__ unsigned lds_off(const void* p){
  return (unsigned)(unsigned long long)p;
}

// 32B tile chunk per lane: two async 16B global->LDS copies. INST_OFFSET is
// added to both the LDS and memory addresses (ISA 08 §4.4), so the second
// chunk is a single offset:16.
__device__ __forceinline__ void async_tile32(unsigned ldsOff, const bf16* src){
  asm volatile("global_load_async_to_lds_b128 %0, %1, off"
               :: "v"(ldsOff), "v"(src) : "memory");
  asm volatile("global_load_async_to_lds_b128 %0, %1, off offset:16"
               :: "v"(ldsOff), "v"(src) : "memory");
}

// ---------------------------------------------------------------------------
// x fp32 -> (copy to d_out, bf16 token-major [BT,D], bf16 atlas-major [A,B,D])
// ---------------------------------------------------------------------------
__global__ __launch_bounds__(256) void convert_x_kernel(
    const float* __restrict__ x, float* __restrict__ xcopy,
    bf16* __restrict__ xtok, bf16* __restrict__ xatl){
  size_t idx = (size_t)blockIdx.x * 256 + threadIdx.x;   // over BT*DIM
  float v = x[idx];
  xcopy[idx] = v;
  bf16 h = f2bf(v);
  xtok[idx] = h;
  size_t t = idx >> 9; unsigned c = (unsigned)(idx & 511);
  size_t b = t >> 4;   unsigned a = (unsigned)(t & 15);
  xatl[(((size_t)a * BATCH + b) << 9) + c] = h;
}

// ---------------------------------------------------------------------------
// W fp32 [K,N] -> bf16 Wt [N,K], batched over grid.z
// ---------------------------------------------------------------------------
__global__ __launch_bounds__(256) void wtrans_kernel(
    const float* __restrict__ src, bf16* __restrict__ dst,
    int K, int N, long long bsS, long long bsD){
  long long g = blockIdx.z;
  size_t idx = (size_t)blockIdx.x * 256 + threadIdx.x;
  if (idx >= (size_t)K * (size_t)N) return;
  int k = (int)(idx / N), n = (int)(idx % N);
  dst[g * bsD + (size_t)n * K + k] = f2bf(src[g * bsS + idx]);
}

// ---------------------------------------------------------------------------
// Grouped WMMA GEMM: C = act(A[M,K] @ W[K,N] + bias), W stored transposed
// [N,K] bf16. Per group g (grid.z): A += g*strideA, W += g*strideW, etc.
// Dual epilogue: fp32 (outF, ldF) and/or bf16 (outB, ldB).
// Block tile 128x128, BK=32, 256 threads = 8 waves, wave tile 32x64.
// M % 128 == 0, N % 128 == 0, K % 32 == 0 assumed (true for all launches).
//
// Double-buffered async pipeline: each wave issues 4 async b128 copies per
// tile; ASYNCcnt completes in order, so after issuing tile t+1's loads,
// s_wait_asynccnt 4 guarantees tile t's writes have landed in LDS. Tile t+1
// streams into the other LDS buffer while WMMAs consume tile t.
// ---------------------------------------------------------------------------
#define BM 128
#define BN 128
#define BK 32
#define LDP 40   // LDS row pitch in bf16 (80B, 16B-aligned chunks)

template<int ACT>  // 0=none, 1=silu, 2=gelu(exact)
__global__ __launch_bounds__(256) void gemm_wmma(
    const bf16* __restrict__ Abase, long long strideA, int lda,
    const bf16* __restrict__ Wbase, long long strideW,
    const float* __restrict__ biasBase, long long strideBias,
    float* __restrict__ outF, long long strideOutF, int ldF,
    bf16* __restrict__ outB, long long strideOutB, int ldB,
    int M, int N, int K){
  __shared__ __align__(16) bf16 lA[2][BM * LDP];
  __shared__ __align__(16) bf16 lW[2][BN * LDP];

  int g = blockIdx.z;
  const bf16*  Ap = Abase + (long long)g * strideA;
  const bf16*  Wp = Wbase + (long long)g * strideW;
  const float* bp = biasBase ? biasBase + (long long)g * strideBias : nullptr;
  float* oF = outF ? outF + (long long)g * strideOutF : nullptr;
  bf16*  oB = outB ? outB + (long long)g * strideOutB : nullptr;

  int bm = blockIdx.y * BM, bn = blockIdx.x * BN;
  int tid  = threadIdx.x;
  int lane = tid & 31, wave = tid >> 5;
  int m0 = (wave & 3) * 32, n0 = (wave >> 2) * 64;
  int fr = lane & 15, kh = lane >> 4, kb8 = kh * 8;
  int lrow = tid >> 1, lch = (tid & 1) * 16;

  const bf16* arow = Ap + (size_t)(bm + lrow) * lda + lch;   // + k0 per tile
  const bf16* wrow = Wp + (size_t)(bn + lrow) * K   + lch;

  unsigned laOff[2] = { lds_off(&lA[0][lrow * LDP + lch]),
                        lds_off(&lA[1][lrow * LDP + lch]) };
  unsigned lwOff[2] = { lds_off(&lW[0][lrow * LDP + lch]),
                        lds_off(&lW[1][lrow * LDP + lch]) };

  f32x8 acc[2][4];
  #pragma unroll
  for (int i = 0; i < 2; i++)
    #pragma unroll
    for (int j = 0; j < 4; j++)
      #pragma unroll
      for (int e = 0; e < 8; e++) acc[i][j][e] = 0.f;

  int nt = K / BK;
  // prologue: tile 0 -> buffer 0
  async_tile32(laOff[0], arow);
  async_tile32(lwOff[0], wrow);

  for (int t = 0; t < nt; t++){
    int cur = t & 1;
    if (t + 1 < nt){
      int k1 = (t + 1) * BK;
      async_tile32(laOff[cur ^ 1], arow + k1);     // tile t+1 -> other buffer
      async_tile32(lwOff[cur ^ 1], wrow + k1);
      if (t + 2 < nt){                              // warm L2 two tiles ahead
        __builtin_prefetch(arow + (t + 2) * BK, 0, 1);   // global_prefetch_b8
        __builtin_prefetch(wrow + (t + 2) * BK, 0, 1);
      }
      // 4 outstanding belong to tile t+1; in-order completion => tile t done.
      asm volatile("s_wait_asynccnt 4" ::: "memory");
    } else {
      asm volatile("s_wait_asynccnt 0" ::: "memory");
    }
    __syncthreads();   // all waves' tile-t LDS writes visible

    Frag fa[2], fb[4];
    #pragma unroll
    for (int i = 0; i < 2; i++){
      int base = (m0 + i * 16 + fr) * LDP + kb8;
      fa[i].u[0] = *(const u32x4*)&lA[cur][base];
      fa[i].u[1] = *(const u32x4*)&lA[cur][base + 16];
    }
    #pragma unroll
    for (int j = 0; j < 4; j++){
      int base = (n0 + j * 16 + fr) * LDP + kb8;
      fb[j].u[0] = *(const u32x4*)&lW[cur][base];
      fb[j].u[1] = *(const u32x4*)&lW[cur][base + 16];
    }
    #pragma unroll
    for (int i = 0; i < 2; i++)
      #pragma unroll
      for (int j = 0; j < 4; j++)
        acc[i][j] = __builtin_amdgcn_wmma_f32_16x16x32_bf16(
            false, fa[i].v, false, fb[j].v, (short)0, acc[i][j], false, false);
    __syncthreads();   // frees buffer cur for tile t+2's async writes
  }

  #pragma unroll
  for (int i = 0; i < 2; i++){
    int rbase = bm + m0 + i * 16 + kh * 8;
    #pragma unroll
    for (int j = 0; j < 4; j++){
      int c = bn + n0 + j * 16 + fr;
      float bv = bp ? bp[c] : 0.f;
      #pragma unroll
      for (int e = 0; e < 8; e++){
        int r = rbase + e;
        float v = acc[i][j][e] + bv;
        if (ACT == 1) v = v / (1.f + __expf(-v));
        else if (ACT == 2) v = 0.5f * v * (1.f + erff(v * 0.70710678118f));
        if (oF) oF[(size_t)r * ldF + c] = v;
        if (oB) oB[(size_t)r * ldB + c] = f2bf(v);
      }
    }
  }
}

// ---------------------------------------------------------------------------
// Differential attention over A=16 atlas tokens. One wave per (head, batch).
// scores via WMMA (q 16x32 x k^T), softmax in LDS, a = p0 - lam*p1 zero-padded
// to 16x32, o = a @ v via 4 WMMAs (v^T staged in LDS), then RMS * 0.2 -> bf16.
// ---------------------------------------------------------------------------
__global__ __launch_bounds__(32) void attn_kernel(
    const bf16* __restrict__ qb, const bf16* __restrict__ kb,
    const bf16* __restrict__ vb,
    const float* __restrict__ lq1, const float* __restrict__ lk1,
    const float* __restrict__ lq2, const float* __restrict__ lk2,
    bf16* __restrict__ obuf){
  int h = blockIdx.x, b = blockIdx.y;
  int lane = threadIdx.x;
  int fr = lane & 15, kh = lane >> 4, kb8 = kh * 8;

  __shared__ float sc[2][16][16];
  __shared__ __align__(16) bf16 aL[16 * 32];
  __shared__ __align__(16) bf16 vT[64 * 32];
  __shared__ float of[16][64];

  size_t tokBase = (size_t)b * 16;

  Frag q0, q1, k0, k1;
  {
    const bf16* qp = qb + (tokBase + fr) * 512 + (2 * h) * 32 + kb8;
    q0.u[0] = *(const u32x4*)qp;        q0.u[1] = *(const u32x4*)(qp + 16);
    q1.u[0] = *(const u32x4*)(qp + 32); q1.u[1] = *(const u32x4*)(qp + 48);
    const bf16* kp = kb + (tokBase + fr) * 512 + (2 * h) * 32 + kb8;
    k0.u[0] = *(const u32x4*)kp;        k0.u[1] = *(const u32x4*)(kp + 16);
    k1.u[0] = *(const u32x4*)(kp + 32); k1.u[1] = *(const u32x4*)(kp + 48);
  }
  f32x8 s0, s1;
  #pragma unroll
  for (int e = 0; e < 8; e++){ s0[e] = 0.f; s1[e] = 0.f; }
  s0 = __builtin_amdgcn_wmma_f32_16x16x32_bf16(false, q0.v, false, k0.v, (short)0, s0, false, false);
  s1 = __builtin_amdgcn_wmma_f32_16x16x32_bf16(false, q1.v, false, k1.v, (short)0, s1, false, false);

  const float scl = 0.17677669529663687f;   // 32^-0.5
  #pragma unroll
  for (int e = 0; e < 8; e++){
    sc[0][kh * 8 + e][fr] = s0[e] * scl;
    sc[1][kh * 8 + e][fr] = s1[e] * scl;
  }
  // stage v^T while scores land: vT[n=feature 0..63][k=token 0..15], pad k>=16
  const bf16 ZB = f2bf(0.f);
  #pragma unroll
  for (int half = 0; half < 2; half++){
    int cc = lane + half * 32;
    for (int j = 0; j < 16; j++)
      vT[cc * 32 + j] = vb[(tokBase + j) * 512 + h * 64 + cc];
    for (int j = 0; j < 16; j++)
      vT[cc * 32 + 16 + j] = ZB;
  }
  __syncthreads();

  float d1 = 0.f, d2 = 0.f;
  for (int t = 0; t < 32; t++){ d1 += lq1[t] * lk1[t]; d2 += lq2[t] * lk2[t]; }
  float lam = __expf(d1) - __expf(d2) + 0.8f;

  // row softmax: lane -> (matrix kh, row fr)
  {
    float* rp = sc[kh][fr];
    float m = rp[0];
    for (int j = 1; j < 16; j++) m = fmaxf(m, rp[j]);
    float sum = 0.f;
    for (int j = 0; j < 16; j++){ float p = __expf(rp[j] - m); rp[j] = p; sum += p; }
    float inv = 1.f / sum;
    for (int j = 0; j < 16; j++) rp[j] *= inv;
  }
  __syncthreads();

  #pragma unroll
  for (int t = 0; t < 8; t++){
    int idx = lane * 8 + t;
    int i = idx >> 4, j = idx & 15;
    aL[i * 32 + j]      = f2bf(sc[0][i][j] - lam * sc[1][i][j]);
    aL[i * 32 + 16 + j] = ZB;
  }
  __syncthreads();

  Frag fa;
  {
    int base = fr * 32 + kb8;
    fa.u[0] = *(const u32x4*)&aL[base];
    fa.u[1] = *(const u32x4*)&aL[base + 16];
  }
  f32x8 o[4];
  #pragma unroll
  for (int j = 0; j < 4; j++){
    #pragma unroll
    for (int e = 0; e < 8; e++) o[j][e] = 0.f;
    Frag fb;
    int base = (j * 16 + fr) * 32 + kb8;
    fb.u[0] = *(const u32x4*)&vT[base];
    fb.u[1] = *(const u32x4*)&vT[base + 16];
    o[j] = __builtin_amdgcn_wmma_f32_16x16x32_bf16(false, fa.v, false, fb.v, (short)0, o[j], false, false);
  }
  #pragma unroll
  for (int j = 0; j < 4; j++)
    #pragma unroll
    for (int e = 0; e < 8; e++)
      of[kh * 8 + e][j * 16 + fr] = o[j][e];
  __syncthreads();

  if (lane < 16){
    float ss = 0.f;
    for (int c = 0; c < 64; c++) ss += of[lane][c] * of[lane][c];
    float r = rsqrtf(ss * (1.f / 64.f) + 1e-5f) * 0.2f;   // rms * (1 - lambda_init)
    for (int c = 0; c < 64; c++)
      obuf[(tokBase + lane) * 512 + h * 64 + c] = f2bf(of[lane][c] * r);
  }
}

// ---------------------------------------------------------------------------
// Router logits + softmax over A=16 -> rw
// ---------------------------------------------------------------------------
__global__ __launch_bounds__(32) void logits_kernel(
    const bf16* __restrict__ o2b, const float* __restrict__ Wl,
    const float* __restrict__ bl, float* __restrict__ rw){
  int b = blockIdx.x, lane = threadIdx.x;
  __shared__ float lg[16];
  if (lane < 16){
    const bf16* p = o2b + ((size_t)b * 16 + lane) * 512;
    float s = 0.f;
    for (int c = 0; c < 512; c++) s += bf2f(p[c]) * Wl[c];
    lg[lane] = s + bl[0];
  }
  __syncthreads();
  if (lane < 16){
    float m = lg[0];
    for (int j = 1; j < 16; j++) m = fmaxf(m, lg[j]);
    float sum = 0.f;
    for (int j = 0; j < 16; j++) sum += __expf(lg[j] - m);
    rw[b * 16 + lane] = __expf(lg[lane] - m) / sum;
  }
}

// ---------------------------------------------------------------------------
// final = [LN(mean_a so), LN(sum_a eo*rw)]
// ---------------------------------------------------------------------------
__global__ __launch_bounds__(256) void final_kernel(
    const float* __restrict__ eo, const float* __restrict__ so,
    const float* __restrict__ rw,
    const float* __restrict__ gsh, const float* __restrict__ bsh,
    const float* __restrict__ gsp, const float* __restrict__ bsp,
    float* __restrict__ fout){
  int b = blockIdx.x, tid = threadIdx.x;
  __shared__ float sv[512], ev[512], red[256], stats[4];
  for (int c = tid; c < 512; c += 256){
    float s = 0.f, e = 0.f;
    for (int a = 0; a < 16; a++){
      size_t base = (((size_t)b * 16 + a) << 9) + c;
      s += so[base];
      e += eo[base] * rw[b * 16 + a];
    }
    sv[c] = s * (1.f / 16.f);
    ev[c] = e;
  }
  __syncthreads();
  float s1 = 0.f, s2 = 0.f, t1 = 0.f, t2 = 0.f;
  for (int c = tid; c < 512; c += 256){
    float v = sv[c]; s1 += v; s2 += v * v;
    float u = ev[c]; t1 += u; t2 += u * u;
  }
  red[tid] = s1; __syncthreads();
  for (int o = 128; o > 0; o >>= 1){ if (tid < o) red[tid] += red[tid + o]; __syncthreads(); }
  if (tid == 0) stats[0] = red[0]; __syncthreads();
  red[tid] = s2; __syncthreads();
  for (int o = 128; o > 0; o >>= 1){ if (tid < o) red[tid] += red[tid + o]; __syncthreads(); }
  if (tid == 0) stats[1] = red[0]; __syncthreads();
  red[tid] = t1; __syncthreads();
  for (int o = 128; o > 0; o >>= 1){ if (tid < o) red[tid] += red[tid + o]; __syncthreads(); }
  if (tid == 0) stats[2] = red[0]; __syncthreads();
  red[tid] = t2; __syncthreads();
  for (int o = 128; o > 0; o >>= 1){ if (tid < o) red[tid] += red[tid + o]; __syncthreads(); }
  if (tid == 0) stats[3] = red[0]; __syncthreads();

  float mS = stats[0] * (1.f / 512.f), vS = stats[1] * (1.f / 512.f) - mS * mS;
  float mE = stats[2] * (1.f / 512.f), vE = stats[3] * (1.f / 512.f) - mE * mE;
  float rS = rsqrtf(vS + 1e-5f), rE = rsqrtf(vE + 1e-5f);
  for (int c = tid; c < 512; c += 256){
    fout[(size_t)b * 1024 + c]       = (sv[c] - mS) * rS * gsh[c] + bsh[c];
    fout[(size_t)b * 1024 + 512 + c] = (ev[c] - mE) * rE * gsp[c] + bsp[c];
  }
}

// ---------------------------------------------------------------------------
extern "C" void kernel_launch(void* const* d_in, const int* in_sizes, int n_in,
                              void* d_out, int out_size, void* d_ws, size_t ws_size,
                              hipStream_t stream) {
  (void)in_sizes; (void)n_in; (void)out_size; (void)ws_size;
  const float* x   = (const float*)d_in[0];
  const float* We1 = (const float*)d_in[1];
  const float* be1 = (const float*)d_in[2];
  const float* We2 = (const float*)d_in[3];
  const float* be2 = (const float*)d_in[4];
  const float* Ws1 = (const float*)d_in[5];
  const float* bs1 = (const float*)d_in[6];
  const float* Ws2 = (const float*)d_in[7];
  const float* bs2 = (const float*)d_in[8];
  const float* Wr1 = (const float*)d_in[9];
  const float* br1 = (const float*)d_in[10];
  const float* Wr2 = (const float*)d_in[11];
  const float* br2 = (const float*)d_in[12];
  const float* g_share = (const float*)d_in[13];
  const float* b_share = (const float*)d_in[14];
  const float* g_spec  = (const float*)d_in[15];
  const float* b_spec  = (const float*)d_in[16];
  const float* Wq = (const float*)d_in[17];
  const float* Wk = (const float*)d_in[18];
  const float* Wv = (const float*)d_in[19];
  const float* Wo = (const float*)d_in[20];
  const float* lq1 = (const float*)d_in[21];
  const float* lk1 = (const float*)d_in[22];
  const float* lq2 = (const float*)d_in[23];
  const float* lk2 = (const float*)d_in[24];
  const float* Wl  = (const float*)d_in[25];
  const float* bl  = (const float*)d_in[26];

  float* out    = (float*)d_out;
  float* o_final= out;
  float* o_eo   = o_final + (size_t)BATCH * DIM2;
  float* o_so   = o_eo  + (size_t)BT * DIM;
  float* o_x    = o_so  + (size_t)BT * DIM;
  float* o_rec  = o_x   + (size_t)BT * DIM;
  float* o_rw   = o_rec + (size_t)BT * DIM;

  char* wsp = (char*)d_ws;
  auto carve = [&](size_t bytes) -> void* {
    void* p = (void*)wsp;
    wsp += (bytes + 255) & ~(size_t)255;
    return p;
  };
  bf16* xtok  = (bf16*)carve((size_t)BT * DIM * 2);
  bf16* xatl  = (bf16*)carve((size_t)BT * DIM * 2);
  bf16* Wqt   = (bf16*)carve((size_t)DIM * DIM * 2);
  bf16* Wkt   = (bf16*)carve((size_t)DIM * DIM * 2);
  bf16* Wvt   = (bf16*)carve((size_t)DIM * DIM * 2);
  bf16* Wot   = (bf16*)carve((size_t)DIM * DIM * 2);
  bf16* We1t  = (bf16*)carve((size_t)ATL * DIM * DIM2 * 2);
  bf16* We2t  = (bf16*)carve((size_t)ATL * DIM2 * DIM * 2);
  bf16* Ws1t  = (bf16*)carve((size_t)DIM * DIM2 * 2);
  bf16* Ws2t  = (bf16*)carve((size_t)DIM2 * DIM * 2);
  bf16* Wr1t  = (bf16*)carve((size_t)DIM2 * DIM * 2);
  bf16* Wr2t  = (bf16*)carve((size_t)DIM * DIM * 2);
  bf16* qb    = (bf16*)carve((size_t)BT * DIM * 2);
  bf16* kbuf  = (bf16*)carve((size_t)BT * DIM * 2);
  bf16* vbuf  = (bf16*)carve((size_t)BT * DIM * 2);
  bf16* obuf  = (bf16*)carve((size_t)BT * DIM * 2);
  bf16* o2b   = (bf16*)carve((size_t)BT * DIM * 2);
  bf16* he    = (bf16*)carve((size_t)ATL * BATCH * DIM2 * 2);
  bf16* hs    = (bf16*)carve((size_t)ATL * BATCH * DIM2 * 2);
  bf16* cat   = (bf16*)carve((size_t)BT * DIM2 * 2);
  bf16* hr    = (bf16*)carve((size_t)BT * DIM * 2);

  dim3 blk(256);

  // x conversion + x passthrough
  convert_x_kernel<<<(unsigned)((size_t)BT * DIM / 256), blk, 0, stream>>>(x, o_x, xtok, xatl);

  // weight transposes (fp32 -> bf16 [N,K])
  wtrans_kernel<<<dim3((DIM * DIM2 + 255) / 256, 1, ATL), blk, 0, stream>>>(We1, We1t, DIM,  DIM2, (long long)DIM * DIM2, (long long)DIM * DIM2);
  wtrans_kernel<<<dim3((DIM2 * DIM + 255) / 256, 1, ATL), blk, 0, stream>>>(We2, We2t, DIM2, DIM,  (long long)DIM2 * DIM, (long long)DIM2 * DIM);
  wtrans_kernel<<<dim3((DIM * DIM2 + 255) / 256, 1, 1),   blk, 0, stream>>>(Ws1, Ws1t, DIM,  DIM2, 0, 0);
  wtrans_kernel<<<dim3((DIM2 * DIM + 255) / 256, 1, 1),   blk, 0, stream>>>(Ws2, Ws2t, DIM2, DIM,  0, 0);
  wtrans_kernel<<<dim3((DIM2 * DIM + 255) / 256, 1, 1),   blk, 0, stream>>>(Wr1, Wr1t, DIM2, DIM,  0, 0);
  wtrans_kernel<<<dim3((DIM * DIM + 255) / 256, 1, 1),    blk, 0, stream>>>(Wr2, Wr2t, DIM,  DIM,  0, 0);
  wtrans_kernel<<<dim3((DIM * DIM + 255) / 256, 1, 1),    blk, 0, stream>>>(Wq,  Wqt,  DIM,  DIM,  0, 0);
  wtrans_kernel<<<dim3((DIM * DIM + 255) / 256, 1, 1),    blk, 0, stream>>>(Wk,  Wkt,  DIM,  DIM,  0, 0);
  wtrans_kernel<<<dim3((DIM * DIM + 255) / 256, 1, 1),    blk, 0, stream>>>(Wv,  Wvt,  DIM,  DIM,  0, 0);
  wtrans_kernel<<<dim3((DIM * DIM + 255) / 256, 1, 1),    blk, 0, stream>>>(Wo,  Wot,  DIM,  DIM,  0, 0);

  // router projections q,k,v : [BT,512] @ [512,512]
  gemm_wmma<0><<<dim3(DIM / BN, BT / BM, 1), blk, 0, stream>>>(
      xtok, 0, DIM, Wqt, 0, nullptr, 0, nullptr, 0, 0, qb,   0, DIM, BT, DIM, DIM);
  gemm_wmma<0><<<dim3(DIM / BN, BT / BM, 1), blk, 0, stream>>>(
      xtok, 0, DIM, Wkt, 0, nullptr, 0, nullptr, 0, 0, kbuf, 0, DIM, BT, DIM, DIM);
  gemm_wmma<0><<<dim3(DIM / BN, BT / BM, 1), blk, 0, stream>>>(
      xtok, 0, DIM, Wvt, 0, nullptr, 0, nullptr, 0, 0, vbuf, 0, DIM, BT, DIM, DIM);

  // differential attention -> obuf (rms'ed, *0.2)
  attn_kernel<<<dim3(HEADS, BATCH), dim3(32), 0, stream>>>(qb, kbuf, vbuf, lq1, lk1, lq2, lk2, obuf);

  // o @ Wo -> o2b; then logits + softmax -> rw
  gemm_wmma<0><<<dim3(DIM / BN, BT / BM, 1), blk, 0, stream>>>(
      obuf, 0, DIM, Wot, 0, nullptr, 0, nullptr, 0, 0, o2b, 0, DIM, BT, DIM, DIM);
  logits_kernel<<<dim3(BATCH), dim3(32), 0, stream>>>(o2b, Wl, bl, o_rw);

  // per-atlas experts: he = silu(x_a @ We1_a + be1_a); eo = he @ We2_a + be2_a
  gemm_wmma<1><<<dim3(DIM2 / BN, BATCH / BM, ATL), blk, 0, stream>>>(
      xatl, (long long)BATCH * DIM, DIM, We1t, (long long)DIM2 * DIM, be1, DIM2,
      nullptr, 0, 0, he, (long long)BATCH * DIM2, DIM2, BATCH, DIM2, DIM);
  gemm_wmma<0><<<dim3(DIM / BN, BATCH / BM, ATL), blk, 0, stream>>>(
      he, (long long)BATCH * DIM2, DIM2, We2t, (long long)DIM * DIM2, be2, DIM,
      o_eo, DIM, ATL * DIM, cat, DIM2, ATL * DIM2, BATCH, DIM, DIM2);

  // shared expert
  gemm_wmma<1><<<dim3(DIM2 / BN, BT / BM, 1), blk, 0, stream>>>(
      xatl, 0, DIM, Ws1t, 0, bs1, 0, nullptr, 0, 0, hs, 0, DIM2, BT, DIM2, DIM);
  gemm_wmma<0><<<dim3(DIM / BN, BATCH / BM, ATL), blk, 0, stream>>>(
      hs, (long long)BATCH * DIM2, DIM2, Ws2t, 0, bs2, 0,
      o_so, DIM, ATL * DIM, cat + DIM, DIM2, ATL * DIM2, BATCH, DIM, DIM2);

  // recon MLP over concat [BT,1024]
  gemm_wmma<2><<<dim3(DIM / BN, BT / BM, 1), blk, 0, stream>>>(
      cat, 0, DIM2, Wr1t, 0, br1, 0, nullptr, 0, 0, hr, 0, DIM, BT, DIM, DIM2);
  gemm_wmma<0><<<dim3(DIM / BN, BT / BM, 1), blk, 0, stream>>>(
      hr, 0, DIM, Wr2t, 0, br2, 0, o_rec, 0, DIM, nullptr, 0, 0, BT, DIM, DIM);

  // final LayerNorms
  final_kernel<<<dim3(BATCH), blk, 0, stream>>>(o_eo, o_so, o_rw, g_share, b_share, g_spec, b_spec, o_final);
}
